// self_LSTM_sparse_one_step_90907277787368
// MI455X (gfx1250) — matrix operand order repaired and checked by reading, not compiled, not run on hardware
//
#include <hip/hip_runtime.h>
#include <hip/hip_bf16.h>
#include <math.h>

// Problem constants (match reference)
#define BATCH 32
#define TSEQ  128
#define DIN   128
#define HDIM  128
#define G4    512      // 4*H
#define TBUF  129      // T+1 history slots
#define NCLS  100
#define TOPK  5
#define EPSV  1e-7f
#define NEGV  -1e9f

typedef __attribute__((ext_vector_type(16))) _Float16 v16h;
typedef __attribute__((ext_vector_type(8)))  _Float16 v8h;
typedef __attribute__((ext_vector_type(8)))  float    v8f;

#define DEV __device__ __forceinline__

DEV float sigmoidf_(float x) { return 1.0f / (1.0f + expf(-x)); }

// A fragment: 16x32 f16 tile, row-major source. Lane (0..15) holds row m=lane&15.
// Per ISA: group g = lane>>4; elements 0..7 -> K = k0 + g*8 + i ; 8..15 -> K = k0+16+g*8+(i-8)
DEV v16h load_a_frag(const _Float16* __restrict__ rowptr, int k0) {
  int lane = threadIdx.x & 31;
  int g = lane >> 4;
  const v8h* p0 = (const v8h*)(rowptr + k0 + g * 8);
  const v8h* p1 = (const v8h*)(rowptr + k0 + 16 + g * 8);
  v8h lo = *p0;
  v8h hi = *p1;
  v16h a;
#pragma unroll
  for (int i = 0; i < 8; ++i) { a[i] = lo[i]; a[8 + i] = hi[i]; }
  return a;
}

// B fragment: 32x16 f16 tile from K-major [K][N] matrix (ldb = N stride).
// Lane holds column N = n0+(lane&15); g = lane>>4; elements 0..7 -> K = k0+g*8+i, 8..15 -> K = k0+16+g*8+(i-8).
DEV v16h load_b_frag(const _Float16* __restrict__ Bm, int ldb, int k0, int n0) {
  int lane = threadIdx.x & 31;
  int g = lane >> 4;
  const _Float16* p = Bm + (size_t)(k0 + g * 8) * ldb + n0 + (lane & 15);
  v16h b;
#pragma unroll
  for (int i = 0; i < 8; ++i) { b[i] = p[(size_t)i * ldb]; b[8 + i] = p[(size_t)(16 + i) * ldb]; }
  return b;
}

DEV v8f wmma_f16(v16h a, v16h b, v8f c) {
  return __builtin_amdgcn_wmma_f32_16x16x32_f16(false, a, false, b, (short)0, c,
                                                false, false);
}

// ---------------------------------------------------------------------------
// K0: one-time prep. f16 conversions, K-major weight transposes, zero state.
// ---------------------------------------------------------------------------
__global__ void __launch_bounds__(256)
k_prep(const float* __restrict__ x, const float* __restrict__ w_ih,
       const float* __restrict__ w_hh, const float* __restrict__ w_t1,
       _Float16* __restrict__ x16, _Float16* __restrict__ wihT,
       _Float16* __restrict__ whhT, _Float16* __restrict__ wt1_16,
       float* __restrict__ h_fin, float* __restrict__ c,
       _Float16* __restrict__ h16, float* __restrict__ buf,
       _Float16* __restrict__ tbuf16, float* __restrict__ attn_c) {
  int stride = gridDim.x * blockDim.x;
  int tid0 = blockIdx.x * blockDim.x + threadIdx.x;
  for (int i = tid0; i < BATCH * TSEQ * DIN; i += stride)
    x16[i] = (_Float16)x[i];
  for (int i = tid0; i < G4 * HDIM; i += stride) {
    int n = i / HDIM, k = i % HDIM;                 // w[n][k] row-major
    wihT[(size_t)k * G4 + n] = (_Float16)w_ih[i];   // -> [K=128][N=512]
    whhT[(size_t)k * G4 + n] = (_Float16)w_hh[i];
  }
  for (int i = tid0; i < 2 * HDIM * HDIM; i += stride)
    wt1_16[i] = (_Float16)w_t1[i];                  // already [K=256][N=128]
  for (int i = tid0; i < BATCH * HDIM; i += stride) {
    h_fin[i] = 0.0f; c[i] = 0.0f; attn_c[i] = 0.0f;
    h16[i] = (_Float16)0.0f;
    int b = i >> 7, j = i & 127;
    buf[((size_t)b * TBUF) * HDIM + j] = 0.0f;          // slot 0 = zero h_old
    tbuf16[((size_t)b * TBUF) * HDIM + j] = (_Float16)0.0f;
  }
}

// ---------------------------------------------------------------------------
// K1: LSTM cell, one step. Single 8-wave block: 64 output tiles (M=32,N=512),
// gates staged in dynamic LDS (64 KB), fused gate activations.
// Writes h_lstm (f32) and tanh(h_lstm) (f16 cache for attention A-matrix).
// ---------------------------------------------------------------------------
__global__ void __launch_bounds__(256)
k_lstm_step(const _Float16* __restrict__ x16, const _Float16* __restrict__ h16,
            const _Float16* __restrict__ wihT, const _Float16* __restrict__ whhT,
            const float* __restrict__ b_ih, const float* __restrict__ b_hh,
            float* __restrict__ c, float* __restrict__ h_lstm,
            _Float16* __restrict__ th16, int t) {
  extern __shared__ float gates[];   // [32][512]
  int tid = threadIdx.x;
  int wave = tid >> 5, lane = tid & 31;
#pragma unroll
  for (int q = 0; q < 8; ++q) {
    int tt = wave + (q << 3);
    int m0 = (tt & 1) * 16;
    int n0 = (tt >> 1) * 16;
    int bi = m0 + (lane & 15);
    const _Float16* arow_x = x16 + ((size_t)bi * TSEQ + t) * DIN;
    const _Float16* arow_h = h16 + (size_t)bi * HDIM;
    v8f acc = {};
#pragma unroll
    for (int ks = 0; ks < 4; ++ks)
      acc = wmma_f16(load_a_frag(arow_x, ks * 32),
                     load_b_frag(wihT, G4, ks * 32, n0), acc);
#pragma unroll
    for (int ks = 0; ks < 4; ++ks)
      acc = wmma_f16(load_a_frag(arow_h, ks * 32),
                     load_b_frag(whhT, G4, ks * 32, n0), acc);
    int nn = n0 + (lane & 15);
    int mbase = m0 + ((lane >> 4) << 3);
#pragma unroll
    for (int r = 0; r < 8; ++r)
      gates[(mbase + r) * G4 + nn] = acc[r];
  }
  __syncthreads();
  for (int e = tid; e < BATCH * HDIM; e += 256) {
    int bi = e >> 7, j = e & 127;
    const float* g = gates + bi * G4;
    float gi = g[j]           + b_ih[j]           + b_hh[j];
    float gf = g[HDIM + j]    + b_ih[HDIM + j]    + b_hh[HDIM + j];
    float gg = g[2*HDIM + j]  + b_ih[2*HDIM + j]  + b_hh[2*HDIM + j];
    float go = g[3*HDIM + j]  + b_ih[3*HDIM + j]  + b_hh[3*HDIM + j];
    float cn = sigmoidf_(gf) * c[e] + sigmoidf_(gi) * tanhf(gg);
    float hn = sigmoidf_(go) * tanhf(cn);
    c[e] = cn;
    h_lstm[e] = hn;
    th16[e] = (_Float16)tanhf(hn);
  }
}

// ---------------------------------------------------------------------------
// K2: attention score GEMM for one step. Rows r = t*32 + b, t < rem.
// grid.x = 2*rem M-tiles; 8 waves = 8 N-tiles; K=256 (tanh(h) ++ tanh(buf)).
// Hidden tanh + dot with w_t_2 fused via shfl + LDS reduction -> scores[b][t].
// ---------------------------------------------------------------------------
__global__ void __launch_bounds__(256)
k_scores(const _Float16* __restrict__ th16, const _Float16* __restrict__ tbuf16,
         const _Float16* __restrict__ wt1, const float* __restrict__ wt2,
         float* __restrict__ scores) {
  __shared__ float red[8][16];
  int tid = threadIdx.x, wave = tid >> 5, lane = tid & 31;
  int m0 = blockIdx.x * 16;
  int r = m0 + (lane & 15);
  int bi = r & 31, ti = r >> 5;
  const _Float16* arow_h = th16 + (size_t)bi * HDIM;
  const _Float16* arow_b = tbuf16 + ((size_t)bi * TBUF + ti) * HDIM;
  int n0 = wave * 16;
  v8f acc = {};
#pragma unroll
  for (int ks = 0; ks < 4; ++ks) {
    __builtin_prefetch(wt1 + (size_t)(128 + ks * 32 + lane) * HDIM + n0, 0, 3);
    acc = wmma_f16(load_a_frag(arow_h, ks * 32),
                   load_b_frag(wt1, HDIM, ks * 32, n0), acc);
  }
#pragma unroll
  for (int ks = 0; ks < 4; ++ks)
    acc = wmma_f16(load_a_frag(arow_b, ks * 32),
                   load_b_frag(wt1, HDIM, 128 + ks * 32, n0), acc);
  int n = n0 + (lane & 15);
  float w2 = wt2[n];
  float pr[8];
#pragma unroll
  for (int rr = 0; rr < 8; ++rr) pr[rr] = tanhf(acc[rr]) * w2;
#pragma unroll
  for (int s = 1; s < 16; s <<= 1) {
#pragma unroll
    for (int rr = 0; rr < 8; ++rr) pr[rr] += __shfl_xor(pr[rr], s, 32);
  }
  if ((lane & 15) == 0) {
    int hs = lane >> 4;
#pragma unroll
    for (int rr = 0; rr < 8; ++rr) red[wave][hs * 8 + rr] = pr[rr];
  }
  __syncthreads();
  if (tid < 16) {
    float s = 0.0f;
#pragma unroll
    for (int w = 0; w < 8; ++w) s += red[w][tid];
    int rg = m0 + tid;
    scores[(rg & 31) * TBUF + (rg >> 5)] = s;
  }
}

// ---------------------------------------------------------------------------
// K3: sparse top-k attention + state update. One wave per batch row.
// ---------------------------------------------------------------------------
__global__ void __launch_bounds__(32)
k_attn(const float* __restrict__ scores, const float* __restrict__ buf,
       const float* __restrict__ h_lstm, float* __restrict__ h_fin,
       _Float16* __restrict__ h16, float* __restrict__ buf_w,
       _Float16* __restrict__ tbuf16, float* __restrict__ attn_c, int rem) {
  int bi = blockIdx.x;
  int lane = threadIdx.x;
  float v[4], wv[4];
#pragma unroll
  for (int q = 0; q < 4; ++q) {
    int t = lane + q * 32;
    v[q] = (t < rem) ? scores[bi * TBUF + t] : NEGV;
  }
  if (rem <= TOPK) {
#pragma unroll
    for (int q = 0; q < 4; ++q) {
      int t = lane + q * 32;
      wv[q] = (t < rem) ? v[q] : 0.0f;   // raw-score branch
    }
  } else {
    float tmp[4] = {v[0], v[1], v[2], v[3]};
    float kth = NEGV;
    for (int it = 0; it < TOPK + 1; ++it) {
      float lm = tmp[0]; int ls = 0;
#pragma unroll
      for (int q = 1; q < 4; ++q)
        if (tmp[q] > lm) { lm = tmp[q]; ls = q; }
      float wm = lm;
#pragma unroll
      for (int s = 1; s < 32; s <<= 1) wm = fmaxf(wm, __shfl_xor(wm, s, 32));
      unsigned long long ball = __ballot(lm == wm);
      int first = __ffsll((long long)ball) - 1;
      if (lane == first) tmp[ls] = NEGV;
      kth = wm;                           // after 6 iters: (TOPK+1)-th largest
    }
    float delta = kth - EPSV;
    float ssum = 0.0f;
#pragma unroll
    for (int q = 0; q < 4; ++q) { wv[q] = fmaxf(v[q] - delta, 0.0f); ssum += wv[q]; }
#pragma unroll
    for (int s = 1; s < 32; s <<= 1) ssum += __shfl_xor(ssum, s, 32);
    float inv = 1.0f / (ssum + EPSV);
#pragma unroll
    for (int q = 0; q < 4; ++q) wv[q] *= inv;
  }
  float acc[4] = {0.0f, 0.0f, 0.0f, 0.0f};
  for (int t = 0; t < rem; ++t) {
    float wt = __shfl(wv[t >> 5], t & 31, 32);   // wave-uniform
    if (wt != 0.0f) {
      const float* bp = buf + ((size_t)bi * TBUF + t) * HDIM;
#pragma unroll
      for (int q = 0; q < 4; ++q) acc[q] += wt * bp[lane + q * 32];
    }
  }
#pragma unroll
  for (int q = 0; q < 4; ++q) {
    int j = lane + q * 32;
    float hf = h_lstm[bi * HDIM + j] + acc[q];
    h_fin[bi * HDIM + j] = hf;
    h16[bi * HDIM + j] = (_Float16)hf;
    attn_c[bi * HDIM + j] = acc[q];
    buf_w[((size_t)bi * TBUF + rem) * HDIM + j] = hf;
    tbuf16[((size_t)bi * TBUF + rem) * HDIM + j] = (_Float16)tanhf(hf);
  }
}

// ---------------------------------------------------------------------------
// K4: final FC [32,256] x [256,100] + bias (tiny; plain VALU).
// ---------------------------------------------------------------------------
__global__ void __launch_bounds__(256)
k_fc(const float* __restrict__ h_fin, const float* __restrict__ attn_c,
     const float* __restrict__ fc_w, const float* __restrict__ fc_b,
     float* __restrict__ out) {
  int idx = blockIdx.x * blockDim.x + threadIdx.x;
  if (idx >= BATCH * NCLS) return;
  int bi = idx / NCLS, n = idx % NCLS;
  const float* w = fc_w + (size_t)n * (2 * HDIM);
  const float* hh = h_fin + (size_t)bi * HDIM;
  const float* ac = attn_c + (size_t)bi * HDIM;
  float s = fc_b[n];
#pragma unroll 4
  for (int k = 0; k < HDIM; ++k) s += hh[k] * w[k] + ac[k] * w[HDIM + k];
  out[idx] = s;
}

// ---------------------------------------------------------------------------
extern "C" void kernel_launch(void* const* d_in, const int* in_sizes, int n_in,
                              void* d_out, int out_size, void* d_ws,
                              size_t ws_size, hipStream_t stream) {
  const float* x    = (const float*)d_in[0];
  const float* w_ih = (const float*)d_in[1];
  const float* w_hh = (const float*)d_in[2];
  const float* b_ih = (const float*)d_in[3];
  const float* b_hh = (const float*)d_in[4];
  const float* w_t1 = (const float*)d_in[5];
  const float* w_t2 = (const float*)d_in[6];
  const float* fc_w = (const float*)d_in[7];
  const float* fc_b = (const float*)d_in[8];
  float* out = (float*)d_out;

  // carve workspace (256B-aligned slabs)
  char* p = (char*)d_ws;
  auto carve = [&](size_t bytes) -> char* {
    char* r = p;
    p += (bytes + 255) & ~(size_t)255;
    return r;
  };
  float*    h_lstm = (float*)carve(BATCH * HDIM * 4);
  float*    h_fin  = (float*)carve(BATCH * HDIM * 4);
  float*    c      = (float*)carve(BATCH * HDIM * 4);
  float*    attn_c = (float*)carve(BATCH * HDIM * 4);
  float*    scores = (float*)carve(BATCH * TBUF * 4);
  _Float16* h16    = (_Float16*)carve(BATCH * HDIM * 2);
  _Float16* th16   = (_Float16*)carve(BATCH * HDIM * 2);
  float*    buf    = (float*)carve((size_t)BATCH * TBUF * HDIM * 4);
  _Float16* tbuf16 = (_Float16*)carve((size_t)BATCH * TBUF * HDIM * 2);
  _Float16* x16    = (_Float16*)carve((size_t)BATCH * TSEQ * DIN * 2);
  _Float16* wihT   = (_Float16*)carve((size_t)HDIM * G4 * 2);
  _Float16* whhT   = (_Float16*)carve((size_t)HDIM * G4 * 2);
  _Float16* wt1_16 = (_Float16*)carve((size_t)2 * HDIM * HDIM * 2);

  k_prep<<<512, 256, 0, stream>>>(x, w_ih, w_hh, w_t1, x16, wihT, whhT, wt1_16,
                                  h_fin, c, h16, buf, tbuf16, attn_c);

  for (int t = 0; t < TSEQ; ++t) {
    int rem = t + 1;
    k_lstm_step<<<1, 256, BATCH * G4 * sizeof(float), stream>>>(
        x16, h16, wihT, whhT, b_ih, b_hh, c, h_lstm, th16, t);
    k_scores<<<2 * rem, 256, 0, stream>>>(th16, tbuf16, wt1_16, w_t2, scores);
    k_attn<<<BATCH, 32, 0, stream>>>(scores, buf, h_lstm, h_fin, h16, buf,
                                     tbuf16, attn_c, rem);
  }

  k_fc<<<(BATCH * NCLS + 255) / 256, 256, 0, stream>>>(h_fin, attn_c, fc_w,
                                                       fc_b, out);
}
